// MultiHeadAttention_22084721836873
// MI455X (gfx1250) — matrix operand order repaired and checked by reading, not compiled, not run on hardware
//
#include <hip/hip_runtime.h>

#define BATCH 4
#define SEQ   1500
#define SEQP  1504
#define HID_  1024
#define NH_   16
#define DH_   64
#define NREL  129     // 2K+1
#define NRELP 160     // padded to 5 x 32 K-chunks
#define MROWS (BATCH*SEQ)   // 6000
#define PBW   132     // padded width for P / wsum LDS rows

typedef __attribute__((ext_vector_type(16))) __bf16 bfvec16;
typedef __attribute__((ext_vector_type(8)))  __bf16 bf16x8;
typedef __attribute__((ext_vector_type(8)))  float  fvec8;
typedef __attribute__((ext_vector_type(4)))  float  f32x4;
typedef __attribute__((ext_vector_type(4)))  unsigned int u32x4;

__device__ __forceinline__ __bf16 tobf(float x) { return (__bf16)x; }

__device__ __forceinline__ void cvt_store8(__bf16* dst, f32x4 a, f32x4 b) {
  bf16x8 r;
  r[0] = (__bf16)a[0]; r[1] = (__bf16)a[1]; r[2] = (__bf16)a[2]; r[3] = (__bf16)a[3];
  r[4] = (__bf16)b[0]; r[5] = (__bf16)b[1]; r[6] = (__bf16)b[2]; r[7] = (__bf16)b[3];
  *(bf16x8*)dst = r;
}

// CDNA5 async global->LDS copy (ASYNCcnt-tracked DMA path, no VGPR round-trip).
// LDS operand = low 32 bits of the generic pointer (== LDS byte offset).
__device__ __forceinline__ void async_load_b128(void* lds, const void* gptr) {
  unsigned l = (unsigned)(unsigned long long)lds;
  asm volatile("global_load_async_to_lds_b128 %0, %1, off" :: "v"(l), "v"(gptr));
}
__device__ __forceinline__ void wait_async0() {   // all async copies done
  asm volatile("s_wait_asynccnt 0x0" ::: "memory");
}
__device__ __forceinline__ void wait_async8() {   // <=8 outstanding (first 8 done)
  asm volatile("s_wait_asynccnt 0x8" ::: "memory");
}

// ---------------------------------------------------------------- prep kernels

// Wq/Wk/Wv [NH,HID,DH] f32 -> bf16 [HID][NH*DH]
__global__ void k_conv_w(const float* __restrict__ w, __bf16* __restrict__ o) {
  int i = blockIdx.x * 256 + threadIdx.x;
  if (i >= NH_ * HID_ * DH_) return;
  int n = i / (HID_ * DH_);
  int r = i - n * HID_ * DH_;
  int h = r / DH_;
  int d = r - h * DH_;
  o[h * HID_ + n * DH_ + d] = tobf(w[i]);
}

__global__ void k_conv_direct(const float* __restrict__ s, __bf16* __restrict__ d, int n) {
  int i = blockIdx.x * 256 + threadIdx.x;
  if (i < n) d[i] = tobf(s[i]);
}

// pos tables [129,64] f32 -> bf16 [160,64] zero padded
__global__ void k_conv_pos(const float* __restrict__ s, __bf16* __restrict__ d) {
  int i = blockIdx.x * 256 + threadIdx.x;
  if (i >= NRELP * DH_) return;
  int r = i / DH_;
  d[i] = (r < NREL) ? tobf(s[i]) : (__bf16)0.f;
}

// zero the padded rows s in [1500,1504) of Q/K/V buffers
__global__ void k_zero_pad(__bf16* qb, __bf16* kb, __bf16* vb) {
  int i = blockIdx.x * 256 + threadIdx.x;
  const int E = BATCH * NH_ * (SEQP - SEQ) * DH_;   // 16384
  if (i >= 3 * E) return;
  int buf = i / E, r = i % E;
  int bn  = r / ((SEQP - SEQ) * DH_);
  int rem = r % ((SEQP - SEQ) * DH_);
  int s = SEQ + rem / DH_;
  int d = rem % DH_;
  __bf16* p = (buf == 0) ? qb : (buf == 1 ? kb : vb);
  p[((size_t)bn * SEQP + s) * DH_ + d] = (__bf16)0.f;
}

// ---------------------------------------------------------------- QKV GEMM
// X [B*S, HID] f32  x  Wb [HID, HID] bf16  ->  Ob [B,NH,SEQP,DH] bf16 (one head per N-tile)
__global__ void __launch_bounds__(128)
k_qkv_gemm(const float* __restrict__ X, const __bf16* __restrict__ Wb,
           const float* __restrict__ bias, __bf16* __restrict__ Ob) {
  __shared__ __align__(16) __bf16 As[64][40];
  __shared__ __align__(16) __bf16 Bs[32][72];
  const int tid = threadIdx.x, lane = tid & 31, wave = tid >> 5;
  const int m0 = blockIdx.x * 64;
  const int head = blockIdx.y;
  const int n0 = head * DH_;

  const int ar = tid >> 1, ac = (tid & 1) * 16;   // A staging: half f32 row / thread
  const int br = tid >> 2, bc = (tid & 3) * 16;   // B staging: quarter bf16 row / thread
  const bool arow_ok = (m0 + ar) < MROWS;
  const float* aptr = X + (size_t)(m0 + ar) * HID_ + ac;

  fvec8 acc[4];
#pragma unroll
  for (int t = 0; t < 4; ++t) acc[t] = fvec8{0.f,0.f,0.f,0.f,0.f,0.f,0.f,0.f};

  for (int kk = 0; kk < HID_; kk += 32) {
    {   // B tile 32x64 bf16: async DMA straight into LDS (overlaps A convert below)
      const __bf16* gp = Wb + (size_t)(kk + br) * HID_ + n0 + bc;
      async_load_b128(&Bs[br][bc], gp);
      async_load_b128(&Bs[br][bc + 8], gp + 8);
    }
    {   // stage A 64x32 f32 -> bf16, vectorized
      f32x4 a0, a1, a2, a3;
      if (arow_ok) {
        const f32x4* p = (const f32x4*)(aptr + kk);
        a0 = p[0]; a1 = p[1]; a2 = p[2]; a3 = p[3];
        if (kk + 32 < HID_) __builtin_prefetch(aptr + kk + 32, 0, 1);
      } else {
        a0 = f32x4{0.f,0.f,0.f,0.f}; a1 = a0; a2 = a0; a3 = a0;
      }
      cvt_store8(&As[ar][ac], a0, a1);
      cvt_store8(&As[ar][ac + 8], a2, a3);
    }
    wait_async0();
    __syncthreads();

    bfvec16 a;
    {
      int M = 16 * wave + (lane & 15);
      int hi = (lane >> 4) * 8;
#pragma unroll
      for (int j = 0; j < 8; ++j) {
        int d0 = 2 * (j & 3) + 16 * (j >> 2) + hi;
        a[2 * j]     = As[M][d0];
        a[2 * j + 1] = As[M][d0 + 1];
      }
    }
#pragma unroll
    for (int t = 0; t < 4; ++t) {
      bfvec16 bf_;
#pragma unroll
      for (int j = 0; j < 8; ++j) {
        int c = 16 * t + (lane & 15);
        int k0 = 2 * (j & 3) + 16 * (j >> 2) + (lane >> 4) * 8;
        bf_[2 * j]     = Bs[k0][c];
        bf_[2 * j + 1] = Bs[k0 + 1][c];
      }
      acc[t] = __builtin_amdgcn_wmma_f32_16x16x32_bf16(false, a, false, bf_,
                                                       (short)0, acc[t], false, false);
    }
    __syncthreads();
  }

  int hi = (lane >> 4) * 8;
#pragma unroll
  for (int t = 0; t < 4; ++t)
#pragma unroll
    for (int j = 0; j < 8; ++j) {
      int r = 16 * wave + j + hi;
      int m = m0 + r;
      if (m >= MROWS) continue;
      int c = 16 * t + (lane & 15);
      int b = m / SEQ, s = m - b * SEQ;
      float vv = acc[t][j] + bias[n0 + c];
      Ob[(((size_t)b * NH_ + head) * SEQP + s) * DH_ + c] = tobf(vv);
    }
}

// ---------------------------------------------------------------- attention
// One block = 32 query rows of one (b,n); 2 waves, 16 rows each. Flash softmax +
// Shaw relative K bias (P = Q*pos_k^T lookup) + relative V bias (wsum * pos_v).
// K/V tiles arrive via async global->LDS DMA: K drained before the score WMMAs,
// V stays in flight through softmax and is drained just before the W*V WMMAs.
__global__ void __launch_bounds__(64)
k_attn(const __bf16* __restrict__ Qb, const __bf16* __restrict__ Kb,
       const __bf16* __restrict__ Vb, const __bf16* __restrict__ Pkb,
       const __bf16* __restrict__ Pvb, __bf16* __restrict__ Hb) {
  __shared__ __align__(16) __bf16 Qa[32][72];
  __shared__ __align__(16) __bf16 Kt[64][72];
  __shared__ __align__(16) __bf16 Vt[64][72];
  __shared__ float  Pb[32][PBW];
  __shared__ float  Ws[32][PBW];
  __shared__ __align__(16) __bf16 Wex[2][16][72];
  __shared__ float  rowf[32];

  const int tid = threadIdx.x, lane = tid & 31, wave = tid >> 5;
  const int q0 = blockIdx.x * 32;
  const int bn = blockIdx.y;                 // b*NH + n
  const size_t base = (size_t)bn * SEQP * DH_;
  const int b = bn >> 4, n = bn & 15;
  const int hhi = (lane >> 4) * 8;
  const int qlocbase = 16 * wave;

  {   // stage Q tile: half row (32 bf16 = 64B) per thread
    int r = tid >> 1, c0 = (tid & 1) * 32;
    const u32x4* p = (const u32x4*)(Qb + base + (size_t)(q0 + r) * DH_ + c0);
    *(u32x4*)&Qa[r][c0]      = p[0];
    *(u32x4*)&Qa[r][c0 + 8]  = p[1];
    *(u32x4*)&Qa[r][c0 + 16] = p[2];
    *(u32x4*)&Qa[r][c0 + 24] = p[3];
  }
  for (int i = tid; i < 32 * PBW; i += 64) Ws[i / PBW][i % PBW] = 0.f;
  __syncthreads();

  bfvec16 qf[2];                              // Q A-fragments, reused everywhere
  {
    int M = 16 * wave + (lane & 15);
#pragma unroll
    for (int kc = 0; kc < 2; ++kc)
#pragma unroll
      for (int j = 0; j < 8; ++j) {
        int d0 = 2 * (j & 3) + 16 * (j >> 2) + hhi + kc * 32;
        qf[kc][2 * j]     = Qa[M][d0];
        qf[kc][2 * j + 1] = Qa[M][d0 + 1];
      }
  }

  // P = Q x pos_k^T : [32][129] relative-key bias table
#pragma unroll
  for (int nt = 0; nt < 9; ++nt) {
    fvec8 pac = fvec8{0.f,0.f,0.f,0.f,0.f,0.f,0.f,0.f};
#pragma unroll
    for (int kc = 0; kc < 2; ++kc) {
      bfvec16 bf_;
      int rr = nt * 16 + (lane & 15);
#pragma unroll
      for (int j = 0; j < 8; ++j) {
        int dd = kc * 32 + 2 * (j & 3) + 16 * (j >> 2) + hhi;
        bf_[2 * j]     = Pkb[rr * DH_ + dd];
        bf_[2 * j + 1] = Pkb[rr * DH_ + dd + 1];
      }
      pac = __builtin_amdgcn_wmma_f32_16x16x32_bf16(false, qf[kc], false, bf_,
                                                    (short)0, pac, false, false);
    }
#pragma unroll
    for (int j = 0; j < 8; ++j) {
      int row = qlocbase + j + hhi;
      int col = nt * 16 + (lane & 15);
      if (col < PBW) Pb[row][col] = pac[j];
    }
  }
  __syncthreads();

  float mrow[8], lrow[8];
#pragma unroll
  for (int j = 0; j < 8; ++j) { mrow[j] = -1e30f; lrow[j] = 0.f; }
  fvec8 o[4];
#pragma unroll
  for (int t = 0; t < 4; ++t) o[t] = fvec8{0.f,0.f,0.f,0.f,0.f,0.f,0.f,0.f};

  for (int kt = 0; kt < (SEQP / 64); ++kt) {
    const int k0 = kt * 64;
    __syncthreads();
    {   // async-DMA K and V rows into LDS: 8 b128 each per thread (one row each)
      int g = k0 + tid;
      if (g < SEQP) {          // wave-uniform (boundary falls on a wave edge)
        const __bf16* gk = Kb + base + (size_t)g * DH_;
        const __bf16* gv = Vb + base + (size_t)g * DH_;
#pragma unroll
        for (int i = 0; i < 8; ++i) async_load_b128(&Kt[tid][i * 8], gk + i * 8);
#pragma unroll
        for (int i = 0; i < 8; ++i) async_load_b128(&Vt[tid][i * 8], gv + i * 8);
      } else {
        u32x4 z = u32x4{0u,0u,0u,0u};
#pragma unroll
        for (int i = 0; i < 8; ++i) { *(u32x4*)&Kt[tid][i * 8] = z; *(u32x4*)&Vt[tid][i * 8] = z; }
      }
    }
    wait_async8();            // K copies (first 8, in order) complete; V still in flight
    __syncthreads();

    float sct[4][8];
#pragma unroll
    for (int nt = 0; nt < 4; ++nt) {
      fvec8 sac = fvec8{0.f,0.f,0.f,0.f,0.f,0.f,0.f,0.f};
#pragma unroll
      for (int kc = 0; kc < 2; ++kc) {
        bfvec16 bf_;
#pragma unroll
        for (int j = 0; j < 8; ++j) {
          int dd = kc * 32 + 2 * (j & 3) + 16 * (j >> 2) + hhi;
          bf_[2 * j]     = Kt[nt * 16 + (lane & 15)][dd];
          bf_[2 * j + 1] = Kt[nt * 16 + (lane & 15)][dd + 1];
        }
        sac = __builtin_amdgcn_wmma_f32_16x16x32_bf16(false, qf[kc], false, bf_,
                                                      (short)0, sac, false, false);
      }
#pragma unroll
      for (int j = 0; j < 8; ++j) {
        int qloc = qlocbase + j + hhi;
        int qg = q0 + qloc;
        int kg = k0 + nt * 16 + (lane & 15);
        float sc;
        if (kg >= SEQ) sc = -1e30f;
        else {
          int rel = kg - qg;
          rel = rel < -64 ? -64 : (rel > 64 ? 64 : rel);
          sc = (sac[j] + Pb[qloc][rel + 64]) * 0.125f;
        }
        sct[nt][j] = sc;
      }
    }

    float scf[8];
#pragma unroll
    for (int j = 0; j < 8; ++j) {
      float t = fmaxf(fmaxf(sct[0][j], sct[1][j]), fmaxf(sct[2][j], sct[3][j]));
#pragma unroll
      for (int mk = 1; mk < 16; mk <<= 1) t = fmaxf(t, __shfl_xor(t, mk, 32));
      float nm = fmaxf(mrow[j], t);
      scf[j] = __expf(mrow[j] - nm);
      mrow[j] = nm;
    }
    if ((lane & 15) == 0)
#pragma unroll
      for (int j = 0; j < 8; ++j) rowf[qlocbase + j + hhi] = scf[j];
    __syncthreads();
    for (int i = lane; i < 16 * PBW; i += 32) {   // rescale this wave's wsum rows
      int r = qlocbase + i / PBW;
      Ws[r][i % PBW] *= rowf[r];
    }

    float rs[8];
#pragma unroll
    for (int j = 0; j < 8; ++j) rs[j] = 0.f;
#pragma unroll
    for (int nt = 0; nt < 4; ++nt)
#pragma unroll
      for (int j = 0; j < 8; ++j) {
        float e = __expf(sct[nt][j] - mrow[j]);
        rs[j] += e;
        int row16 = j + hhi;
        int col = nt * 16 + (lane & 15);
        Wex[wave][row16][col] = tobf(e);
        int qg = q0 + qlocbase + row16;
        int kg = k0 + col;
        if (kg < SEQ) {
          int rel = kg - qg;
          rel = rel < -64 ? -64 : (rel > 64 ? 64 : rel);
          atomicAdd(&Ws[qlocbase + row16][rel + 64], e);
        }
      }
#pragma unroll
    for (int j = 0; j < 8; ++j) {
      float t = rs[j];
#pragma unroll
      for (int mk = 1; mk < 16; mk <<= 1) t += __shfl_xor(t, mk, 32);
      lrow[j] = lrow[j] * scf[j] + t;
    }
#pragma unroll
    for (int t4 = 0; t4 < 4; ++t4)
#pragma unroll
      for (int j = 0; j < 8; ++j) o[t4][j] *= scf[j];
    wait_async0();            // V copies complete (overlapped with all of the above)
    __syncthreads();

    bfvec16 af[2];                              // exp-weights as A fragments
    {
      int M = lane & 15;
#pragma unroll
      for (int kc = 0; kc < 2; ++kc)
#pragma unroll
        for (int j = 0; j < 8; ++j) {
          int c0 = 2 * (j & 3) + 16 * (j >> 2) + hhi + kc * 32;
          af[kc][2 * j]     = Wex[wave][M][c0];
          af[kc][2 * j + 1] = Wex[wave][M][c0 + 1];
        }
    }
#pragma unroll
    for (int td = 0; td < 4; ++td)
#pragma unroll
      for (int kc = 0; kc < 2; ++kc) {
        bfvec16 bf_;
#pragma unroll
        for (int j = 0; j < 8; ++j) {
          int dcol = td * 16 + (lane & 15);
          int kr = kc * 32 + 2 * (j & 3) + 16 * (j >> 2) + hhi;
          bf_[2 * j]     = Vt[kr][dcol];
          bf_[2 * j + 1] = Vt[kr + 1][dcol];
        }
        o[td] = __builtin_amdgcn_wmma_f32_16x16x32_bf16(false, af[kc], false, bf_,
                                                        (short)0, o[td], false, false);
      }
  }

  // finalize: normalize O and wsum, then O += wsum_norm x pos_v
  float inv[8];
#pragma unroll
  for (int j = 0; j < 8; ++j) inv[j] = 1.f / lrow[j];
#pragma unroll
  for (int t4 = 0; t4 < 4; ++t4)
#pragma unroll
    for (int j = 0; j < 8; ++j) o[t4][j] *= inv[j];
  if ((lane & 15) == 0)
#pragma unroll
    for (int j = 0; j < 8; ++j) rowf[qlocbase + j + hhi] = inv[j];
  __syncthreads();
  for (int i = lane; i < 16 * PBW; i += 32) {
    int r = qlocbase + i / PBW;
    Ws[r][i % PBW] *= rowf[r];
  }
  __syncthreads();

#pragma unroll
  for (int kc = 0; kc < 5; ++kc) {              // K = 129 padded to 160
    bfvec16 af;
    int M = qlocbase + (lane & 15);
#pragma unroll
    for (int j = 0; j < 8; ++j) {
      int c0 = 2 * (j & 3) + 16 * (j >> 2) + hhi + kc * 32;
      af[2 * j]     = (c0     < NREL) ? tobf(Ws[M][c0])     : (__bf16)0.f;
      af[2 * j + 1] = (c0 + 1 < NREL) ? tobf(Ws[M][c0 + 1]) : (__bf16)0.f;
    }
#pragma unroll
    for (int td = 0; td < 4; ++td) {
      bfvec16 bf_;
      int dcol = td * 16 + (lane & 15);
#pragma unroll
      for (int j = 0; j < 8; ++j) {
        int rr = kc * 32 + 2 * (j & 3) + 16 * (j >> 2) + hhi;
        bf_[2 * j]     = Pvb[rr * DH_ + dcol];
        bf_[2 * j + 1] = Pvb[(rr + 1) * DH_ + dcol];
      }
      o[td] = __builtin_amdgcn_wmma_f32_16x16x32_bf16(false, af, false, bf_,
                                                      (short)0, o[td], false, false);
    }
  }

#pragma unroll
  for (int td = 0; td < 4; ++td)
#pragma unroll
    for (int j = 0; j < 8; ++j) {
      int qloc = qlocbase + j + hhi;
      int s = q0 + qloc;
      if (s >= SEQ) continue;
      int d = td * 16 + (lane & 15);
      Hb[((size_t)(b * SEQ + s)) * HID_ + n * DH_ + d] = tobf(o[td][j]);
    }
}

// ---------------------------------------------------------------- output FC
__global__ void __launch_bounds__(128)
k_fc_gemm(const __bf16* __restrict__ Hb, const __bf16* __restrict__ Wb,
          const float* __restrict__ bias, float* __restrict__ Out) {
  __shared__ __align__(16) __bf16 As[64][40];
  __shared__ __align__(16) __bf16 Bs[32][72];
  const int tid = threadIdx.x, lane = tid & 31, wave = tid >> 5;
  const int m0 = blockIdx.x * 64;
  const int n0 = blockIdx.y * 64;

  const int ar = tid >> 1, ac = (tid & 1) * 16;
  const int br = tid >> 2, bc = (tid & 3) * 16;
  const bool arow_ok = (m0 + ar) < MROWS;
  const __bf16* aptr = Hb + (size_t)(m0 + ar) * HID_ + ac;

  fvec8 acc[4];
#pragma unroll
  for (int t = 0; t < 4; ++t) acc[t] = fvec8{0.f,0.f,0.f,0.f,0.f,0.f,0.f,0.f};

  for (int kk = 0; kk < HID_; kk += 32) {
    {   // B tile via async DMA
      const __bf16* gp = Wb + (size_t)(kk + br) * HID_ + n0 + bc;
      async_load_b128(&Bs[br][bc], gp);
      async_load_b128(&Bs[br][bc + 8], gp + 8);
    }
    {   // A tile (bf16) vectorized through VGPRs
      u32x4 a0 = u32x4{0u,0u,0u,0u}, a1 = a0;
      if (arow_ok) {
        const u32x4* p = (const u32x4*)(aptr + kk);
        a0 = p[0]; a1 = p[1];
        if (kk + 32 < HID_) __builtin_prefetch(aptr + kk + 32, 0, 1);
      }
      *(u32x4*)&As[ar][ac]     = a0;
      *(u32x4*)&As[ar][ac + 8] = a1;
    }
    wait_async0();
    __syncthreads();
    bfvec16 a;
    {
      int M = 16 * wave + (lane & 15);
      int hi = (lane >> 4) * 8;
#pragma unroll
      for (int j = 0; j < 8; ++j) {
        int d0 = 2 * (j & 3) + 16 * (j >> 2) + hi;
        a[2 * j]     = As[M][d0];
        a[2 * j + 1] = As[M][d0 + 1];
      }
    }
#pragma unroll
    for (int t = 0; t < 4; ++t) {
      bfvec16 bf_;
#pragma unroll
      for (int j = 0; j < 8; ++j) {
        int c = 16 * t + (lane & 15);
        int k0 = 2 * (j & 3) + 16 * (j >> 2) + (lane >> 4) * 8;
        bf_[2 * j]     = Bs[k0][c];
        bf_[2 * j + 1] = Bs[k0 + 1][c];
      }
      acc[t] = __builtin_amdgcn_wmma_f32_16x16x32_bf16(false, a, false, bf_,
                                                       (short)0, acc[t], false, false);
    }
    __syncthreads();
  }
  int hi = (lane >> 4) * 8;
#pragma unroll
  for (int t = 0; t < 4; ++t)
#pragma unroll
    for (int j = 0; j < 8; ++j) {
      int r = 16 * wave + j + hi;
      int m = m0 + r;
      if (m >= MROWS) continue;
      int c = 16 * t + (lane & 15);
      Out[(size_t)m * HID_ + n0 + c] = acc[t][j] + bias[n0 + c];
    }
}

// ---------------------------------------------------------------- launch
extern "C" void kernel_launch(void* const* d_in, const int* in_sizes, int n_in,
                              void* d_out, int out_size, void* d_ws, size_t ws_size,
                              hipStream_t stream) {
  const float* q    = (const float*)d_in[0];
  const float* k    = (const float*)d_in[1];
  const float* v    = (const float*)d_in[2];
  const float* Wq   = (const float*)d_in[3];
  const float* bq   = (const float*)d_in[4];
  const float* Wk   = (const float*)d_in[5];
  const float* bk   = (const float*)d_in[6];
  const float* Wv   = (const float*)d_in[7];
  const float* bv   = (const float*)d_in[8];
  const float* posk = (const float*)d_in[9];
  const float* posv = (const float*)d_in[10];
  const float* Wfc  = (const float*)d_in[11];
  const float* bfc  = (const float*)d_in[12];
  float* out = (float*)d_out;

  char* ws = (char*)d_ws;
  size_t off = 0;
  auto alloc = [&](size_t bytes) -> void* {
    void* p = ws + off;
    off = (off + bytes + 255) & ~(size_t)255;
    return p;
  };
  const size_t QKVE = (size_t)BATCH * NH_ * SEQP * DH_;
  __bf16* qb    = (__bf16*)alloc(QKVE * 2);
  __bf16* kb    = (__bf16*)alloc(QKVE * 2);
  __bf16* vb    = (__bf16*)alloc(QKVE * 2);
  __bf16* wqb   = (__bf16*)alloc((size_t)HID_ * HID_ * 2);
  __bf16* wkb   = (__bf16*)alloc((size_t)HID_ * HID_ * 2);
  __bf16* wvb   = (__bf16*)alloc((size_t)HID_ * HID_ * 2);
  __bf16* wfcb  = (__bf16*)alloc((size_t)HID_ * HID_ * 2);
  __bf16* poskb = (__bf16*)alloc((size_t)NRELP * DH_ * 2);
  __bf16* posvb = (__bf16*)alloc((size_t)NRELP * DH_ * 2);
  __bf16* hb    = (__bf16*)alloc((size_t)6016 * HID_ * 2);

  const int WN = NH_ * HID_ * DH_;
  k_conv_w<<<(WN + 255) / 256, 256, 0, stream>>>(Wq, wqb);
  k_conv_w<<<(WN + 255) / 256, 256, 0, stream>>>(Wk, wkb);
  k_conv_w<<<(WN + 255) / 256, 256, 0, stream>>>(Wv, wvb);
  k_conv_direct<<<(HID_ * HID_ + 255) / 256, 256, 0, stream>>>(Wfc, wfcb, HID_ * HID_);
  k_conv_pos<<<(NRELP * DH_ + 255) / 256, 256, 0, stream>>>(posk, poskb);
  k_conv_pos<<<(NRELP * DH_ + 255) / 256, 256, 0, stream>>>(posv, posvb);
  k_zero_pad<<<(3 * BATCH * NH_ * (SEQP - SEQ) * DH_ + 255) / 256, 256, 0, stream>>>(qb, kb, vb);

  dim3 gg((MROWS + 63) / 64, NH_);
  k_qkv_gemm<<<gg, 128, 0, stream>>>(q, wqb, bq, qb);
  k_qkv_gemm<<<gg, 128, 0, stream>>>(k, wkb, bk, kb);
  k_qkv_gemm<<<gg, 128, 0, stream>>>(v, wvb, bv, vb);

  dim3 ga((SEQ + 31) / 32, BATCH * NH_);
  k_attn<<<ga, 64, 0, stream>>>(qb, kb, vb, poskb, posvb, hb);

  dim3 gf((MROWS + 63) / 64, HID_ / 64);
  k_fc_gemm<<<gf, 128, 0, stream>>>(hb, wfcb, bfc, out);

  (void)in_sizes; (void)n_in; (void)out_size; (void)ws_size;
}
